// MultiHeadSpatialAttention_28424093564999
// MI455X (gfx1250) — matrix operand order repaired and checked by reading, not compiled, not run on hardware
//
#include <hip/hip_runtime.h>
#include <hip/hip_bf16.h>

// ---------------------------------------------------------------------------
// MultiHeadSpatialAttention for MI455X (gfx1250, wave32, WMMA f16 16x16x32)
//
// B=2, C=512, W=H=64 (N=4096), HEADS=4, DH=32, HIDDEN=128.
// Returns (out[2,512,64,64], attn[2,4,4096,4096]) concatenated in d_out.
//
// HBM roofline: ~22 GFLOP of math vs a mandatory 536MB fp32 attn write
// (~23us at 23.3 TB/s). Design: touch HBM once for attn, keep K/V (4MB f16)
// in L2, and spend free WMMA flops to eliminate LDS/barrier serialization.
//
// Head-fuse folded into Q:  fused_g = (wf[g,h] * q_h) concat over h  @  K_all^T
// -> 4 chained v_wmma_f32_16x16x32_f16 per 16x16 score tile (K-dim 128),
//    NO cross-wave exchange, NO barriers in the inner loops.
//
//  K1 qkv_kernel : fp32 GEMM 128x512 -> q/k/v, f16 operand layouts:
//                  Q16[b][h][n][32] (pre-scaled by 1/sqrt(128)),
//                  K16[b][h][n][32], Vt[b*128+hid][n] (transposed for PV-B).
//  K2 attn_kernel: 1 block = 16 query rows x 4 waves (wave = fused head g).
//                  Two-pass online softmax (shfl_xor butterflies); pass 2
//                  writes attn once and feeds P into 2 PV WMMAs via a
//                  per-wave padded-LDS C->A transpose (in-order, barrier-free).
//  K3 outproj    : 512x128 projection + bias + eval-mode BatchNorm.
// ---------------------------------------------------------------------------

typedef __attribute__((ext_vector_type(16))) _Float16 v16h;
typedef __attribute__((ext_vector_type(8)))  float    v8f;

#define HEADS 4
#define DH    32
#define HIDDEN 128
#define CCH   512
#define NPIX  4096
#define BATCH 2
#define INV_TEMP 0.08838834764831845f   // 1/sqrt(128)

// -------------------------------- K1: QKV ---------------------------------
__global__ __launch_bounds__(256) void qkv_kernel(
    const float* __restrict__ x,
    const float* __restrict__ wq, const float* __restrict__ bq,
    const float* __restrict__ wk, const float* __restrict__ bk,
    const float* __restrict__ wv, const float* __restrict__ bv,
    _Float16* __restrict__ Q16, _Float16* __restrict__ K16,
    _Float16* __restrict__ Vt)
{
    const int n = blockIdx.x * blockDim.x + threadIdx.x;   // pixel
    const int o = blockIdx.y;                              // hidden channel
    const int b = blockIdx.z;

    const float* xb  = x  + (size_t)b * CCH * NPIX + n;
    const float* wqp = wq + (size_t)o * CCH;
    const float* wkp = wk + (size_t)o * CCH;
    const float* wvp = wv + (size_t)o * CCH;

    float aq = 0.f, ak = 0.f, av = 0.f;
    #pragma unroll 4
    for (int c = 0; c < CCH; ++c) {
        float xv = xb[(size_t)c * NPIX];           // coalesced across lanes
        aq = fmaf(wqp[c], xv, aq);                 // scalar (uniform) weights
        ak = fmaf(wkp[c], xv, ak);
        av = fmaf(wvp[c], xv, av);
    }
    const int h = o >> 5, d = o & 31;
    const size_t rowMajor = (((size_t)(b * HEADS + h) * NPIX) + n) * DH + d;
    Q16[rowMajor] = (_Float16)((aq + bq[o]) * INV_TEMP);
    K16[rowMajor] = (_Float16)(ak + bk[o]);
    Vt[((size_t)(b * HIDDEN + o)) * NPIX + n] = (_Float16)(av + bv[o]);
}

// ------------------------------ K2: attention ------------------------------
__global__ __launch_bounds__(128) void attn_kernel(
    const _Float16* __restrict__ Q16, const _Float16* __restrict__ K16,
    const _Float16* __restrict__ Vt,  const float* __restrict__ w_fuse,
    float* __restrict__ attn_out, float* __restrict__ acc_out)
{
    const int b    = blockIdx.y;
    const int n0   = blockIdx.x * 16;          // 16 query rows per block
    const int wave = threadIdx.x >> 5;         // wave == fused output head g
    const int lane = threadIdx.x & 31;
    const int lg   = lane >> 4;                // lane group (0/1)
    const int lm   = lane & 15;

    // per-wave staging for P (C-layout -> A-layout), no cross-wave sharing:
    // same-wave LDS ops are in-order, so no barriers are needed at all.
    __shared__ _Float16 sP[HEADS][16][34];

    union V16 { v16h v; unsigned u[8]; };

    // --- A operands: Q'_g = wf[g,h] * Q_h, one 16x32 f16 A-tile per head h.
    //     (fuse is linear & pre-softmax, so it commutes into Q; reduction
    //      dim becomes (h,d)=128 -> 4 chained WMMAs per score tile.)
    V16 qm[HEADS];
    #pragma unroll
    for (int h = 0; h < HEADS; ++h) {
        const _Float16* qp =
            Q16 + (((size_t)(b * HEADS + h) * NPIX) + n0 + lm) * DH;
        #pragma unroll
        for (int r = 0; r < 8; ++r) {
            const int base = ((r < 4) ? 2 * r : 16 + 2 * (r - 4)) + lg * 8;
            qm[h].u[r] = *(const unsigned*)(qp + base);
        }
        const _Float16 wh = (_Float16)w_fuse[wave * 4 + h];
        #pragma unroll
        for (int e = 0; e < 16; ++e)           // packed f16 muls, one-time
            qm[h].v[e] = qm[h].v[e] * wh;
    }

    const _Float16* Kb = K16 + ((size_t)b * HEADS * NPIX) * DH;

    // fused 16x16 score tile at key column mt: 4 chained WMMAs over heads
    auto score_tile = [&](int mt) -> v8f {
        v8f s = {};
        #pragma unroll
        for (int h = 0; h < HEADS; ++h) {
            V16 kb;                             // B operand: K_h^T 32x16
            const _Float16* kp = Kb + (((size_t)h * NPIX) + mt + lm) * DH;
            #pragma unroll
            for (int r = 0; r < 8; ++r)
                kb.u[r] = *(const unsigned*)(kp + ((r < 4) ? 2 * r : 16 + 2 * (r - 4)) + lg * 8);
            s = __builtin_amdgcn_wmma_f32_16x16x32_f16(
                    false, qm[h].v, false, kb.v, (short)0, s, false, false);
        }
        return s;
    };

    // row-statistics update (16-lane butterfly max + sum) for one subtile
    auto stat_update = [&](const v8f& f, float rmax[8], float rsum[8]) {
        #pragma unroll
        for (int i = 0; i < 8; ++i) {
            float tmax = f[i];
            #pragma unroll
            for (int off = 1; off < 16; off <<= 1)
                tmax = fmaxf(tmax, __shfl_xor(tmax, off, 32));
            const float nm = fmaxf(rmax[i], tmax);
            float ts = __expf(f[i] - nm);
            #pragma unroll
            for (int off = 1; off < 16; off <<= 1)
                ts += __shfl_xor(ts, off, 32);
            rsum[i] = rsum[i] * __expf(rmax[i] - nm) + ts;
            rmax[i] = nm;
        }
    };

    // prefetch all heads' K rows for the next chunk (L2-resident stream)
    auto prefetch_chunk = [&](int m0) {
        const int mpf = (m0 + 32) & (NPIX - 1);
        #pragma unroll
        for (int h = 0; h < HEADS; ++h)
            __builtin_prefetch(Kb + (((size_t)h * NPIX) + mpf + lm) * DH, 0, 3);
    };

    // ---------------- pass 1: online row max / normalizer -------------------
    float rmax[8], rsum[8];
    #pragma unroll
    for (int i = 0; i < 8; ++i) { rmax[i] = -3.0e38f; rsum[i] = 0.f; }

    for (int m0 = 0; m0 < NPIX; m0 += 32) {
        prefetch_chunk(m0);
        v8f f0 = score_tile(m0);
        v8f f1 = score_tile(m0 + 16);
        stat_update(f0, rmax, rsum);
        stat_update(f1, rmax, rsum);
    }
    float rinv[8];
    #pragma unroll
    for (int i = 0; i < 8; ++i) rinv[i] = 1.0f / rsum[i];

    // -------- pass 2: normalize, emit attn once, accumulate P @ V ----------
    v8f accD[2]; accD[0] = (v8f){}; accD[1] = (v8f){};
    float* attn_row =
        attn_out + (((size_t)(b * HEADS + wave) * NPIX) + n0) * NPIX;
    const _Float16* Vb = Vt + ((size_t)(b * HIDDEN + wave * DH)) * NPIX;

    for (int m0 = 0; m0 < NPIX; m0 += 32) {     // 32 keys == PV WMMA K-dim
        prefetch_chunk(m0);
        v8f f0 = score_tile(m0);
        v8f f1 = score_tile(m0 + 16);
        #pragma unroll
        for (int i = 0; i < 8; ++i) {
            const int row = i + 8 * lg;
            const float p0 = __expf(f0[i] - rmax[i]) * rinv[i];
            const float p1 = __expf(f1[i] - rmax[i]) * rinv[i];
            attn_row[(size_t)row * NPIX + m0 + lm]      = p0;
            attn_row[(size_t)row * NPIX + m0 + 16 + lm] = p1;
            sP[wave][row][lm]      = (_Float16)p0;      // C->A staging
            sP[wave][row][16 + lm] = (_Float16)p1;
        }
        // same-wave LDS RAW: hardware keeps DS ops in order, no barrier.
        V16 pa;                                  // A operand: P 16x32
        #pragma unroll
        for (int r = 0; r < 8; ++r) {
            const int base = ((r < 4) ? 2 * r : 16 + 2 * (r - 4)) + lg * 8;
            pa.u[r] = *(const unsigned*)&sP[wave][lm][base];
        }
        #pragma unroll
        for (int dh = 0; dh < 2; ++dh) {
            V16 vb;                              // B operand: V 32x16 (from Vt)
            const _Float16* vp = Vb + ((size_t)(dh * 16 + lm)) * NPIX + m0;
            #pragma unroll
            for (int r = 0; r < 8; ++r)
                vb.u[r] = *(const unsigned*)(vp + ((r < 4) ? 2 * r : 16 + 2 * (r - 4)) + lg * 8);
            accD[dh] = __builtin_amdgcn_wmma_f32_16x16x32_f16(
                false, pa.v, false, vb.v, (short)0, accD[dh], false, false);
        }
    }

    // out_acc stored [b][hid][n] so the projection kernel reads coalesced in n
    #pragma unroll
    for (int dh = 0; dh < 2; ++dh)
        #pragma unroll
        for (int i = 0; i < 8; ++i)
            acc_out[((size_t)(b * HIDDEN + wave * DH + dh * 16 + lm)) * NPIX +
                    n0 + i + 8 * lg] = accD[dh][i];
}

// --------------------------- K3: out proj + BN -----------------------------
__global__ __launch_bounds__(256) void outproj_kernel(
    const float* __restrict__ acc,  const float* __restrict__ w_out,
    const float* __restrict__ b_out,
    const float* __restrict__ bn_gamma, const float* __restrict__ bn_beta,
    const float* __restrict__ bn_mean,  const float* __restrict__ bn_var,
    float* __restrict__ out)
{
    const int n = blockIdx.x * blockDim.x + threadIdx.x;
    const int o = blockIdx.y;
    const int b = blockIdx.z;

    const float* ap = acc + (size_t)b * HIDDEN * NPIX + n;
    const float* wp = w_out + (size_t)o * HIDDEN;
    float s = 0.f;
    #pragma unroll 8
    for (int h = 0; h < HIDDEN; ++h)
        s = fmaf(wp[h], ap[(size_t)h * NPIX], s);
    const float y = s + b_out[o];
    const float g = bn_gamma[o] * rsqrtf(bn_var[o] + 1e-5f);
    out[((size_t)b * CCH + o) * NPIX + n] = (y - bn_mean[o]) * g + bn_beta[o];
}

// ------------------------------- launcher ----------------------------------
extern "C" void kernel_launch(void* const* d_in, const int* in_sizes, int n_in,
                              void* d_out, int out_size, void* d_ws, size_t ws_size,
                              hipStream_t stream) {
    (void)in_sizes; (void)n_in; (void)out_size; (void)ws_size;
    const float* x      = (const float*)d_in[0];
    const float* wq     = (const float*)d_in[1];
    const float* bq     = (const float*)d_in[2];
    const float* wk     = (const float*)d_in[3];
    const float* bk     = (const float*)d_in[4];
    const float* wv     = (const float*)d_in[5];
    const float* bv     = (const float*)d_in[6];
    const float* w_fuse = (const float*)d_in[7];
    const float* w_out  = (const float*)d_in[8];
    const float* b_out  = (const float*)d_in[9];
    const float* gamma  = (const float*)d_in[10];
    const float* beta   = (const float*)d_in[11];
    const float* mean   = (const float*)d_in[12];
    const float* var    = (const float*)d_in[13];

    const size_t qkvElems = (size_t)BATCH * HEADS * NPIX * DH;   // 1M halves
    _Float16* Q16 = (_Float16*)d_ws;
    _Float16* K16 = Q16 + qkvElems;
    _Float16* Vt  = K16 + qkvElems;
    float*    acc = (float*)(Vt + qkvElems);                     // 4 MB

    float* out_main = (float*)d_out;
    float* attn     = out_main + (size_t)BATCH * CCH * NPIX;     // tuple part 2

    qkv_kernel<<<dim3(NPIX / 256, HIDDEN, BATCH), 256, 0, stream>>>(
        x, wq, bq, wk, bk, wv, bv, Q16, K16, Vt);

    attn_kernel<<<dim3(NPIX / 16, BATCH), 128, 0, stream>>>(
        Q16, K16, Vt, w_fuse, attn, acc);

    outproj_kernel<<<dim3(NPIX / 256, CCH, BATCH), 256, 0, stream>>>(
        acc, w_out, b_out, gamma, beta, mean, var, out_main);
}
